// CardAwarePolicy_20796231647916
// MI455X (gfx1250) — matrix operand order repaired and checked by reading, not compiled, not run on hardware
//
#include <hip/hip_runtime.h>
#include <hip/hip_bf16.h>

typedef _Float16 half_t;
typedef __attribute__((ext_vector_type(16))) _Float16 v16h;
typedef __attribute__((ext_vector_type(8)))  float    v8f;
typedef __attribute__((ext_vector_type(4)))  unsigned uv4;

#define HS_    8
#define E_     12
#define A_ACT  20
#define TILE_M 64
#define NWAVES 2

// ---- workspace byte offsets (total 92768 B) ----
#define WS_ABIAS 0        // float[20*32]     2560 B
#define WS_BGD   2560     // float[128]        512 B  (concat(gs_b1, dp_b1))
#define WS_W1P   3072     // half [128*64]   16384 B  (ctx_w1, K padded 36->64)
#define WS_W2H   19456    // half [128*128]  32768 B  (ctx_w2)
#define WS_W3H   52224    // half [32*128]    8192 B  (sc_w1[:, :128])
#define WS_WGD   60416    // half [128*96]   24576 B  (blockdiag(gs_w1, dp_w1))
#define WS_QKV   84992    // float[54*36]     7776 B  (per-card QKV projections)

// ---- LDS map (51712 B) with lifetime-based aliasing ----
// [0     .. 8192 ) sX    half[64][64]   (P1 out -> G1 in)
// [8192  .. 24576) region A: sHGD half[64][128] (G0 out -> P1 in)
//                          then sH1  half[64][128] (G1 out -> G2 in)
// [24576 .. 40960) region B: sGD half[64][96] (P0 out -> G0 in)
//                          then sCTX half[64][128] (G2 out -> G3 in)
// [40960 .. 49152) region C: sQKV float[54*36] (P0 out -> P1 in)
//                          then sH0 float[64][32] (G3 out -> P2 in)
// [49152 .. 51712) sAB float[640]
#define SM_X    0
#define SM_A    8192
#define SM_B    24576
#define SM_C    40960
#define SM_AB   49152
#define SM_TOT  51712

// ============ WMMA fragment helpers (CDNA5 16x16x32 f16, wave32) ============
// A 16x32 f16: lane&15 = row M; lane>>4 selects K-halves. VGPR i(0..3): K=h*8+2i,
// VGPR 4+i: K=16+h*8+2i  -> two 16B LDS loads per fragment.
__device__ __forceinline__ v16h load_frag_A(const half_t* base, int ld, int m0,
                                            int k0, int lane) {
  int m = m0 + (lane & 15);
  int h = lane >> 4;
  const half_t* row = base + m * ld + k0 + h * 8;
  union { v16h v; uv4 q[2]; } r;
  r.q[0] = *(const uv4*)(row);        // K = k0+h*8    .. +7
  r.q[1] = *(const uv4*)(row + 16);   // K = k0+16+h*8 .. +7
  return r.v;
}

// B 32x16 f16 (K x N): lane&15 = col N; lane>>4 selects K-halves. VGPR i(0..3): K=h*8+2i, VGPR 4+i: K=16+h*8+2i.
// W row-major [N][K] -> B[k][n]=W[n][k]; two 8-half runs at K=h*8 and K=16+h*8 -> two b128 loads.
__device__ __forceinline__ v16h load_frag_B(const half_t* W, int ld, int n0,
                                            int k0, int lane) {
  int n = n0 + (lane & 15);
  int h = lane >> 4;
  const half_t* row = W + n * ld + k0 + h * 8;
  union { v16h v; uv4 q[2]; } r;
  r.q[0] = *(const uv4*)(row);
  r.q[1] = *(const uv4*)(row + 16);
  return r.v;
}

// D f32 16x16: VGPR r, lane l -> row = r + 8*(l>>4), col = l&15
__device__ __forceinline__ void store_D_half(half_t* dst, int ld, int m0, int n0,
                                             int lane, v8f acc,
                                             const float* __restrict__ bias,
                                             bool relu) {
  int n = n0 + (lane & 15);
  int m = m0 + ((lane >> 4) << 3);
  float bn = bias ? bias[n] : 0.f;
#pragma unroll
  for (int r = 0; r < 8; ++r) {
    float x = acc[r] + bn;
    if (relu) x = fmaxf(x, 0.f);
    dst[(m + r) * ld + n] = (half_t)x;
  }
}

__device__ __forceinline__ void store_D_f32(float* dst, int ld, int m0, int n0,
                                            int lane, v8f acc) {
  int n = n0 + (lane & 15);
  int m = m0 + ((lane >> 4) << 3);
#pragma unroll
  for (int r = 0; r < 8; ++r) dst[(m + r) * ld + n] = acc[r];
}

// =========================== prep kernels ===========================
__global__ void prep_abias(const int* __restrict__ aci,
                           const float* __restrict__ card_emb,
                           const float* __restrict__ sc_w1,
                           const float* __restrict__ sc_b1,
                           float* __restrict__ abias) {
  int t = threadIdx.x;
  if (t >= A_ACT * 32) return;
  int a = t >> 5, j = t & 31;
  float arep[E_];
#pragma unroll
  for (int e = 0; e < E_; ++e) arep[e] = 0.f;
  float cnt = 0.f;
#pragma unroll
  for (int c = 0; c < 4; ++c) {
    int idx = aci[a * 4 + c];
    if (idx != 0) {
      cnt += 1.f;
#pragma unroll
      for (int e = 0; e < E_; ++e) arep[e] += card_emb[idx * E_ + e];
    }
  }
  float inv = 1.f / fmaxf(cnt, 1.f);
  float acc = sc_b1[j];
#pragma unroll
  for (int e = 0; e < E_; ++e) acc += arep[e] * inv * sc_w1[j * 140 + 128 + e];
  abias[a * 32 + j] = acc;
}

// Per-card QKV projection: qkvg[card][o] = in_b[o] + card_emb[card,:] . in_w[o,:]
// (the attention input depends only on card identity -> 54x36 table)
__global__ void prep_qkv(const float* __restrict__ card_emb,
                         const float* __restrict__ in_w,
                         const float* __restrict__ in_b,
                         float* __restrict__ qkvg) {
  int i = blockIdx.x * blockDim.x + threadIdx.x;
  if (i >= 54 * 36) return;
  int card = i / 36, o = i % 36;
  float acc = in_b[o];
#pragma unroll
  for (int c = 0; c < E_; ++c) acc += card_emb[card * E_ + c] * in_w[o * E_ + c];
  qkvg[i] = acc;
}

__global__ void prep_weights(const float* __restrict__ ctx_w1,
                             const float* __restrict__ ctx_w2,
                             const float* __restrict__ sc_w1,
                             const float* __restrict__ gs_w1,
                             const float* __restrict__ dp_w1,
                             const float* __restrict__ gs_b1,
                             const float* __restrict__ dp_b1,
                             half_t* __restrict__ W1p, half_t* __restrict__ W2h,
                             half_t* __restrict__ W3h, half_t* __restrict__ Wgd,
                             float* __restrict__ bgd) {
  const int N1 = 128 * 64;
  const int N2 = N1 + 128 * 128;
  const int N3 = N2 + 32 * 128;
  const int N4 = N3 + 128 * 96;
  const int N5 = N4 + 128;
  int i = blockIdx.x * blockDim.x + threadIdx.x;
  if (i < N1) {                          // ctx_w1 (128x36) -> padded (128x64)
    int n = i >> 6, k = i & 63;
    W1p[i] = (half_t)(k < 36 ? ctx_w1[n * 36 + k] : 0.f);
  } else if (i < N2) {                   // ctx_w2 (128x128)
    int j = i - N1;
    W2h[j] = (half_t)ctx_w2[j];
  } else if (i < N3) {                   // sc_w1[:, :128]
    int j = i - N2;
    int n = j >> 7, k = j & 127;
    W3h[j] = (half_t)sc_w1[n * 140 + k];
  } else if (i < N4) {                   // blockdiag(gs_w1[64x12], dp_w1[64x54])
    int j = i - N3;
    int n = j / 96, k = j % 96;
    float v = 0.f;
    if (n < 64) {
      if (k < 12) v = gs_w1[n * 12 + k];
    } else {
      int kk = k - 12;
      if (kk >= 0 && kk < 54) v = dp_w1[(n - 64) * 54 + kk];
    }
    Wgd[j] = (half_t)v;
  } else if (i < N5) {                   // concat biases
    int n = i - N4;
    bgd[n] = (n < 64) ? gs_b1[n] : dp_b1[n - 64];
  }
}

// =========================== main fused kernel ===========================
__global__ __launch_bounds__(TILE_M) void policy_main(
    const int* __restrict__ hand_cards, const float* __restrict__ game_state,
    const float* __restrict__ discard, const int* __restrict__ enemy_card,
    const int* __restrict__ hand_size, const int* __restrict__ nva_p,
    const float* __restrict__ enemy_emb,
    const float* __restrict__ out_w, const float* __restrict__ out_b,
    const float* __restrict__ gs_w2, const float* __restrict__ gs_b2,
    const float* __restrict__ dp_w2, const float* __restrict__ dp_b2,
    const float* __restrict__ ctx_b1, const float* __restrict__ ctx_b2,
    const float* __restrict__ sc_w2, const float* __restrict__ sc_b2,
    const float* __restrict__ abias, const float* __restrict__ bgd,
    const float* __restrict__ qkvg,
    const half_t* __restrict__ W1p, const half_t* __restrict__ W2h,
    const half_t* __restrict__ W3h, const half_t* __restrict__ Wgd,
    float* __restrict__ out, int Btot) {
  __shared__ __align__(16) unsigned char smem[SM_TOT];
  half_t* sX   = (half_t*)(smem + SM_X);   // [64][64]
  half_t* sHGD = (half_t*)(smem + SM_A);   // [64][128] (phase G0 out)
  half_t* sH1  = (half_t*)(smem + SM_A);   // [64][128] (phase G1 out)
  half_t* sGD  = (half_t*)(smem + SM_B);   // [64][96]  (phase P0 out)
  half_t* sCTX = (half_t*)(smem + SM_B);   // [64][128] (phase G2 out)
  float*  sQKV = (float*)(smem + SM_C);    // [54][36]  (P0 out -> P1 in)
  float*  sH0  = (float*)(smem + SM_C);    // [64][32]  (G3 out -> P2 in)
  float*  sAB  = (float*)(smem + SM_AB);   // [20][32]

  const int t    = threadIdx.x;
  const int lane = t & 31;
  const int wv   = t >> 5;
  const int s    = blockIdx.x * TILE_M + t;
  const bool active = s < Btot;

  // warm L2 with the shared f16 weights (global_prefetch_b8 path)
  __builtin_prefetch(W2h + (t << 6), 0, 1);
  __builtin_prefetch(Wgd + (t << 6), 0, 1);
  __builtin_prefetch(W1p + (t << 4), 0, 1);

  // ---- P0: stage abias, QKV table, and [game_state|discard] rows ----------
  for (int i = t; i < A_ACT * 32; i += TILE_M) sAB[i] = abias[i];
  for (int i = t; i < 54 * 36; i += TILE_M) sQKV[i] = qkvg[i];
  if (active) {
#pragma unroll
    for (int c = 0; c < 12; ++c)
      sGD[t * 96 + c] = (half_t)game_state[(size_t)s * 12 + c];
#pragma unroll
    for (int c = 0; c < 54; ++c)
      sGD[t * 96 + 12 + c] = (half_t)discard[(size_t)s * 54 + c];
  } else {
#pragma unroll
    for (int c = 0; c < 66; ++c) sGD[t * 96 + c] = (half_t)0.f;
  }
#pragma unroll
  for (int c = 66; c < 96; ++c) sGD[t * 96 + c] = (half_t)0.f;
  __syncthreads();

  // ---- G0: HGD = relu([gs|dp] @ blockdiag(gs_w1,dp_w1)^T + [gs_b1|dp_b1]) ---
  for (int tile = wv; tile < 4 * 8; tile += NWAVES) {
    int m0 = (tile >> 3) << 4;
    int n0 = (tile & 7) << 4;
    v8f acc = {};
#pragma unroll
    for (int k = 0; k < 96; k += 32) {
      v16h a = load_frag_A(sGD, 96, m0, k, lane);
      v16h b = load_frag_B(Wgd, 96, n0, k, lane);
      acc = __builtin_amdgcn_wmma_f32_16x16x32_f16(false, a, false, b,
                                                   (short)0, acc, false, false);
    }
    store_D_half(sHGD, 128, m0, n0, lane, acc, bgd, true);
  }
  __syncthreads();

  // ---- P1: per-sample attention (table gathers) + MLP 2nd layers -> X row --
  float xrow[36];
  if (active) {
    int cards[HS_];
#pragma unroll
    for (int j = 0; j < HS_; ++j) cards[j] = hand_cards[(size_t)s * HS_ + j];
    float xacc[E_];
#pragma unroll
    for (int i = 0; i < E_; ++i) xacc[i] = 0.f;
    const float scl = 0.57735026919f;  // 1/sqrt(HD)
#pragma clang loop unroll(disable)
    for (int h = 0; h < 4; ++h) {
      // gather per-head K/V for the 8 hand cards from the LDS table
      float kk[HS_][3], vv[HS_][3];
#pragma unroll
      for (int j = 0; j < HS_; ++j) {
        const float* qr = sQKV + cards[j] * 36;
#pragma unroll
        for (int dd = 0; dd < 3; ++dd) {
          kk[j][dd] = qr[12 + h * 3 + dd];
          vv[j][dd] = qr[24 + h * 3 + dd];
        }
      }
      float ah0 = 0.f, ah1 = 0.f, ah2 = 0.f;  // sum over queries of att output
#pragma unroll
      for (int qi = 0; qi < HS_; ++qi) {
        const float* qq = sQKV + cards[qi] * 36 + h * 3;
        float q0 = qq[0], q1 = qq[1], q2 = qq[2];
        float sc[HS_], mx = -3.0e38f;
#pragma unroll
        for (int j = 0; j < HS_; ++j) {
          float d0 = q0 * kk[j][0] + q1 * kk[j][1] + q2 * kk[j][2];
          sc[j] = (cards[j] != 0) ? d0 * scl : -1.0e9f;
          mx = fmaxf(mx, sc[j]);
        }
        float sum = 0.f;
#pragma unroll
        for (int j = 0; j < HS_; ++j) {
          sc[j] = __expf(sc[j] - mx);
          sum += sc[j];
        }
        float inv = 1.f / sum;
        float a0 = 0.f, a1 = 0.f, a2 = 0.f;
#pragma unroll
        for (int j = 0; j < HS_; ++j) {
          a0 += sc[j] * vv[j][0];
          a1 += sc[j] * vv[j][1];
          a2 += sc[j] * vv[j][2];
        }
        ah0 += a0 * inv;
        ah1 += a1 * inv;
        ah2 += a2 * inv;
      }
      // fold this head's slice through the output projection (h only in
      // global addresses -> no dynamic register-array indexing anywhere)
      const float* wo = out_w + h * 3;
#pragma unroll
      for (int o = 0; o < E_; ++o)
        xacc[o] += ah0 * wo[o * E_] + ah1 * wo[o * E_ + 1] + ah2 * wo[o * E_ + 2];
    }
    float invl = 1.f / fmaxf((float)hand_size[s], 1.f);
#pragma unroll
    for (int o = 0; o < E_; ++o) xrow[o] = (xacc[o] + 8.f * out_b[o]) * invl;
    // enemy embedding
    int ec = enemy_card[s];
#pragma unroll
    for (int c = 0; c < E_; ++c) xrow[12 + c] = enemy_emb[ec * E_ + c];
    // MLP second layers: read WMMA-produced hidden row (limit load hoisting)
    const half_t* hrow = sHGD + t * 128;
    float g6[6], d6[6];
#pragma unroll
    for (int c = 0; c < 6; ++c) g6[c] = gs_b2[c];
#pragma unroll
    for (int c = 0; c < 6; ++c) d6[c] = dp_b2[c];
#pragma unroll 8
    for (int j = 0; j < 64; ++j) {
      float hg = (float)hrow[j];
#pragma unroll
      for (int c = 0; c < 6; ++c) g6[c] += gs_w2[c * 64 + j] * hg;
    }
#pragma unroll 8
    for (int j = 0; j < 64; ++j) {
      float hd = (float)hrow[64 + j];
#pragma unroll
      for (int c = 0; c < 6; ++c) d6[c] += dp_w2[c * 64 + j] * hd;
    }
#pragma unroll
    for (int c = 0; c < 6; ++c) xrow[24 + c] = g6[c];
#pragma unroll
    for (int c = 0; c < 6; ++c) xrow[30 + c] = d6[c];
  } else {
#pragma unroll
    for (int c = 0; c < 36; ++c) xrow[c] = 0.f;
  }
  // write padded f16 X row
#pragma unroll
  for (int c = 0; c < 36; ++c) sX[t * 64 + c] = (half_t)xrow[c];
#pragma unroll
  for (int c = 36; c < 64; ++c) sX[t * 64 + c] = (half_t)0.f;
  __syncthreads();

  // ---- G1: H1 = relu(X @ W1p^T + b1)  (64x128, K=64) ----------------------
  for (int tile = wv; tile < 4 * 8; tile += NWAVES) {
    int m0 = (tile >> 3) << 4;
    int n0 = (tile & 7) << 4;
    v8f acc = {};
#pragma unroll
    for (int k = 0; k < 64; k += 32) {
      v16h a = load_frag_A(sX, 64, m0, k, lane);
      v16h b = load_frag_B(W1p, 64, n0, k, lane);
      acc = __builtin_amdgcn_wmma_f32_16x16x32_f16(false, a, false, b,
                                                   (short)0, acc, false, false);
    }
    store_D_half(sH1, 128, m0, n0, lane, acc, ctx_b1, true);
  }
  __syncthreads();

  // ---- G2: CTX = H1 @ W2^T + b2  (64x128, K=128) --------------------------
  for (int tile = wv; tile < 4 * 8; tile += NWAVES) {
    int m0 = (tile >> 3) << 4;
    int n0 = (tile & 7) << 4;
    v8f acc = {};
#pragma unroll
    for (int k = 0; k < 128; k += 32) {
      v16h a = load_frag_A(sH1, 128, m0, k, lane);
      v16h b = load_frag_B(W2h, 128, n0, k, lane);
      acc = __builtin_amdgcn_wmma_f32_16x16x32_f16(false, a, false, b,
                                                   (short)0, acc, false, false);
    }
    store_D_half(sCTX, 128, m0, n0, lane, acc, ctx_b2, false);
  }
  __syncthreads();

  // ---- G3: H0 = CTX @ W3^T  (64x32, K=128, f32 out) -----------------------
  for (int tile = wv; tile < 4 * 2; tile += NWAVES) {
    int m0 = (tile >> 1) << 4;
    int n0 = (tile & 1) << 4;
    v8f acc = {};
#pragma unroll
    for (int k = 0; k < 128; k += 32) {
      v16h a = load_frag_A(sCTX, 128, m0, k, lane);
      v16h b = load_frag_B(W3h, 128, n0, k, lane);
      acc = __builtin_amdgcn_wmma_f32_16x16x32_f16(false, a, false, b,
                                                   (short)0, acc, false, false);
    }
    store_D_f32(sH0, 32, m0, n0, lane, acc);
  }
  __syncthreads();

  // ---- P2: scorer (algebraically reduced: relu(h0 + abias[a]) . sc_w2) ----
  if (active) {
    float h0r[32], w2[32];
#pragma unroll
    for (int j = 0; j < 32; ++j) h0r[j] = sH0[t * 32 + j];
#pragma unroll
    for (int j = 0; j < 32; ++j) w2[j] = sc_w2[j];
    float b2 = sc_b2[0];
    int nv = nva_p[0];
#pragma clang loop unroll(disable)
    for (int a = 0; a < A_ACT; ++a) {
      float acc = b2;
      const float* ab = sAB + a * 32;
#pragma unroll
      for (int j = 0; j < 32; ++j)
        acc += fmaxf(h0r[j] + ab[j], 0.f) * w2[j];
      out[(size_t)s * A_ACT + a] = (a < nv) ? acc : -1.0e8f;
    }
  }
}

// =========================== launcher ===========================
extern "C" void kernel_launch(void* const* d_in, const int* in_sizes, int n_in,
                              void* d_out, int out_size, void* d_ws,
                              size_t ws_size, hipStream_t stream) {
  const int*   hand_cards = (const int*)d_in[0];
  const float* game_state = (const float*)d_in[1];
  const float* discard    = (const float*)d_in[2];
  const int*   enemy_card = (const int*)d_in[3];
  const int*   hand_size  = (const int*)d_in[4];
  const int*   aci        = (const int*)d_in[5];
  const int*   nva        = (const int*)d_in[6];
  const float* card_emb   = (const float*)d_in[7];
  const float* enemy_emb  = (const float*)d_in[8];
  const float* in_w  = (const float*)d_in[9];
  const float* in_b  = (const float*)d_in[10];
  const float* out_w = (const float*)d_in[11];
  const float* out_b = (const float*)d_in[12];
  const float* gs_w1 = (const float*)d_in[13];
  const float* gs_b1 = (const float*)d_in[14];
  const float* gs_w2 = (const float*)d_in[15];
  const float* gs_b2 = (const float*)d_in[16];
  const float* dp_w1 = (const float*)d_in[17];
  const float* dp_b1 = (const float*)d_in[18];
  const float* dp_w2 = (const float*)d_in[19];
  const float* dp_b2 = (const float*)d_in[20];
  const float* ctx_w1 = (const float*)d_in[21];
  const float* ctx_b1 = (const float*)d_in[22];
  const float* ctx_w2 = (const float*)d_in[23];
  const float* ctx_b2 = (const float*)d_in[24];
  const float* sc_w1 = (const float*)d_in[25];
  const float* sc_b1 = (const float*)d_in[26];
  const float* sc_w2 = (const float*)d_in[27];
  const float* sc_b2 = (const float*)d_in[28];

  char* ws = (char*)d_ws;
  float*  abias = (float*)(ws + WS_ABIAS);
  float*  bgd   = (float*)(ws + WS_BGD);
  half_t* W1p   = (half_t*)(ws + WS_W1P);
  half_t* W2h   = (half_t*)(ws + WS_W2H);
  half_t* W3h   = (half_t*)(ws + WS_W3H);
  half_t* Wgd   = (half_t*)(ws + WS_WGD);
  float*  qkvg  = (float*)(ws + WS_QKV);

  int Btot = in_sizes[0] / HS_;

  prep_abias<<<1, A_ACT * 32, 0, stream>>>(aci, card_emb, sc_w1, sc_b1, abias);
  prep_qkv<<<(54 * 36 + 255) / 256, 256, 0, stream>>>(card_emb, in_w, in_b,
                                                      qkvg);
  int wele = 128 * 64 + 128 * 128 + 32 * 128 + 128 * 96 + 128;
  prep_weights<<<(wele + 255) / 256, 256, 0, stream>>>(
      ctx_w1, ctx_w2, sc_w1, gs_w1, dp_w1, gs_b1, dp_b1, W1p, W2h, W3h, Wgd,
      bgd);
  int nblk = (Btot + TILE_M - 1) / TILE_M;
  policy_main<<<nblk, TILE_M, 0, stream>>>(
      hand_cards, game_state, discard, enemy_card, hand_size, nva, enemy_emb,
      out_w, out_b, gs_w2, gs_b2, dp_w2, dp_b2, ctx_b1, ctx_b2, sc_w2, sc_b2,
      abias, bgd, qkvg, W1p, W2h, W3h, Wgd, (float*)d_out, Btot);
}
